// memes_ED_norm_16501264351917
// MI455X (gfx1250) — matrix operand order repaired and checked by reading, not compiled, not run on hardware
//
#include <hip/hip_runtime.h>

typedef __attribute__((ext_vector_type(16))) __bf16          v16bf;
typedef __attribute__((ext_vector_type(16))) unsigned short  v16us;
typedef __attribute__((ext_vector_type(8)))  float           v8f;
typedef __attribute__((ext_vector_type(4)))  unsigned int    v4u;
typedef __attribute__((ext_vector_type(8)))  unsigned int    v8u;

#define NB    32     // batch
#define NC    3      // in channels
#define HIN   224
#define WIN   224
#define NK    64     // templates
#define HO    218
#define WO    218
#define NQ    21     // filter rows = c*7+r pairs
#define KP    192    // k' padded: 21 rows * 8 (s padded 7->8), rounded to 6 chunks of 32
#define NKC   6      // k-chunks of 32
#define TH    4      // output rows per block
#define TW    64     // output cols per block
#define SLABW 72     // slab row stride (cols 0..71 all written, finite)

static __device__ __forceinline__ unsigned short f2bf(float f) {
    unsigned int u = __builtin_bit_cast(unsigned int, f);
    return (unsigned short)((u + 0x8000u) >> 16);
}
static __device__ __forceinline__ float bf2f(unsigned short h) {
    unsigned int u = ((unsigned int)h) << 16;
    return __builtin_bit_cast(float, u);
}

__global__ __launch_bounds__(256)
void memes_ed_norm_kernel(const float* __restrict__ x,
                          const float* __restrict__ memes,
                          float* __restrict__ out)
{
    __shared__ unsigned short Wlds[NK * KP];        // 24576 B: weights, s-padded, zeros in pads
    __shared__ unsigned short Xlds[NC * 10 * SLABW];// 4320 B : input slab, every col finite
    __shared__ unsigned short Xim[TW * KP];         // 24576 B: im2col rows for current hh
    __shared__ float          X2[TH * TW];
    __shared__ float          M2[NK];
    __shared__ float          Minv[NK];

    const int tid = threadIdx.x;
    const int w0  = blockIdx.x * TW;
    const int h0  = blockIdx.y * TH;
    const int b   = blockIdx.z;

    // ---- weights -> LDS bf16 in padded k' layout: k' = (c*7+r)*8 + s, zero at s==7 / q>=21
    for (int i = tid; i < NK * KP; i += 256) {
        int n = i / KP, kp = i - n * KP;
        int q = kp >> 3, s = kp & 7;
        Wlds[i] = (q < NQ && s < 7) ? f2bf(memes[n * 147 + q * 7 + s]) : (unsigned short)0;
    }
    // ---- input slab -> LDS: 3 ch x 10 rows x 72 cols, ALL cols written (finite; clamp->0)
    for (int i = tid; i < NC * 10 * SLABW; i += 256) {
        int c = i / (10 * SLABW), rem = i - c * (10 * SLABW);
        int rr = rem / SLABW, cc = rem - rr * SLABW;
        int h = h0 + rr, w = w0 + cc;
        float v = (h < HIN && w < WIN) ? x[((b * NC + c) * HIN + h) * WIN + w] : 0.0f;
        Xlds[(c * 10 + rr) * SLABW + cc] = f2bf(v);
    }
    // ---- m2 and 1/m2 in full f32 precision (from global weights)
    if (tid < NK) {
        float s = 0.f;
        for (int k = 0; k < 147; ++k) { float wv = memes[tid * 147 + k]; s += wv * wv; }
        M2[tid] = s; Minv[tid] = 1.0f / s;
    }
    __syncthreads();

    // ---- per-position patch energy x2 (one (hh, p) per thread), from slab
    {
        int hh = tid >> 6, p = tid & 63;
        float s = 0.f;
        for (int c = 0; c < NC; ++c)
            #pragma unroll
            for (int r = 0; r < 7; ++r) {
                int base = (c * 10 + r + hh) * SLABW + p;
                #pragma unroll
                for (int s7 = 0; s7 < 7; ++s7) {
                    float v = bf2f(Xlds[base + s7]);
                    s += v * v;
                }
            }
        X2[tid] = s;
    }

    // ---- wave work assignment: 8 waves = 4 template-tiles x 2 position-tile pairs
    const int wave  = tid >> 5, lane = tid & 31;
    const int mt    = wave & 3;
    const int ntA   = (wave >> 2) * 2;
    const int laneM = lane & 15, hi = lane >> 4;

    // ---- preload A fragments (weights): per chunk, two contiguous 16B runs
    //      lane covers k' in [kc*32+hi*16, +16)
    v16bf A[NKC];
    {
        const unsigned short* wrow = &Wlds[(mt * 16 + laneM) * KP];
        #pragma unroll
        for (int kc = 0; kc < NKC; ++kc) {
            v8u w;
            w.lo = *(const v4u*)(wrow + kc * 32 + hi * 16);
            w.hi = *(const v4u*)(wrow + kc * 32 + hi * 16 + 8);
            A[kc] = __builtin_bit_cast(v16bf, w);
        }
    }

    const int p0 = ntA * 16 + laneM;           // first position this lane covers
    const unsigned short* xr0 = &Xim[p0 * KP]; // 384B-aligned rows
    const unsigned short* xr1 = xr0 + 16 * KP;

    for (int hh = 0; hh < TH; ++hh) {
        // ---- produce im2col rows for this output row: one 16B store per (p, q) task
        const int rowoff = hh * SLABW;
        for (int j = tid; j < TW * 24; j += 256) {
            int q = j >> 6, p = j & 63;
            union { unsigned short u[8]; v4u v; } pk;
            if (q < NQ) {
                int c = q / 7, r = q - c * 7;
                int src = (c * 10 + r) * SLABW + rowoff + p;
                #pragma unroll
                for (int s = 0; s < 8; ++s) pk.u[s] = Xlds[src + s]; // s==7 finite junk: weight=0
            } else {
                pk.v = (v4u)(0u);                                    // q-pad rows: zeros
            }
            *(v4u*)&Xim[p * KP + q * 8] = pk.v;
        }
        __syncthreads();

        // ---- GEMM: D(16 templates x 16 positions) accumulate over 6 chunks of 32
        v8f acc0 = {}; v8f acc1 = {};
        #pragma unroll
        for (int kc = 0; kc < NKC; ++kc) {
            // B fragment: lane = position, 32 contiguous bytes of k' (32B-aligned)
            v8u b0 = *(const v8u*)(xr0 + kc * 32 + hi * 16);
            v8u b1 = *(const v8u*)(xr1 + kc * 32 + hi * 16);
            v16bf B0 = __builtin_bit_cast(v16bf, b0);
            v16bf B1 = __builtin_bit_cast(v16bf, b1);
            acc0 = __builtin_amdgcn_wmma_f32_16x16x32_bf16(
                       false, A[kc], false, B0, (short)0, acc0, false, false);
            acc1 = __builtin_amdgcn_wmma_f32_16x16x32_bf16(
                       false, A[kc], false, B1, (short)0, acc1, false, false);
        }

        // ---- epilogue: lanes = consecutive w_out (coalesced), VGPR r = template row.
        //      Output is a ~389MB stream (>2x L2): use non-temporal stores so it does
        //      not evict the L2-hot weights/input lines shared by all 7040 blocks.
        const int h_out = h0 + hh;
        if (h_out < HO) {
            #pragma unroll
            for (int t = 0; t < 2; ++t) {
                int p = (ntA + t) * 16 + laneM;
                int w_out = w0 + p;
                if (w_out < WO) {
                    float x2v = X2[hh * 64 + p];
                    v8f acc = t ? acc1 : acc0;
                    #pragma unroll
                    for (int r = 0; r < 8; ++r) {
                        int n_g = mt * 16 + hi * 8 + r;
                        float val = (x2v - 2.0f * acc[r] + M2[n_g]) * Minv[n_g];
                        __builtin_nontemporal_store(
                            val, &out[((b * NK + n_g) * HO + h_out) * HO + w_out]);
                    }
                }
            }
        }
        __syncthreads();   // Xim consumed; safe to overwrite next iteration
    }
}

extern "C" void kernel_launch(void* const* d_in, const int* in_sizes, int n_in,
                              void* d_out, int out_size, void* d_ws, size_t ws_size,
                              hipStream_t stream) {
    const float* x     = (const float*)d_in[0];   // [32,3,224,224] f32
    const float* memes = (const float*)d_in[1];   // [64,3,7,7] f32
    float* out = (float*)d_out;                   // [32,64,218,218] f32

    dim3 grid((WO + TW - 1) / TW,   // 4 position tiles
              (HO + TH - 1) / TH,   // 55 row tiles
              NB);                  // 32 batches
    memes_ed_norm_kernel<<<grid, dim3(256), 0, stream>>>(x, memes, out);
}
